// WASMOptimizedTransformer_28630251995540
// MI455X (gfx1250) — matrix operand order repaired and checked by reading, not compiled, not run on hardware
//
#include <hip/hip_runtime.h>
#include <hip/hip_bf16.h>
#include <math.h>

// ---------------- model constants ----------------
#define Dm   1024
#define Hh   8
#define DKk  128
#define Ss   512
#define Bb   8
#define Ll   4
#define FFf  4096
#define NTOK 4096      // B*S

// GEMM tiling: block = 256 thr = 8 waves (2x4 wave grid), wave tile 32x32 (2x2 WMMA)
#define BM   64
#define BN   128
#define BK   32
#define BSTR 136       // padded LDS row stride in halves (128 + 4 DW pad, 16B multiple)

#ifndef USE_TDM
#define USE_TDM 1      // Tensor Data Mover staging for B tiles
#endif

typedef __attribute__((ext_vector_type(16))) _Float16 v16h;
typedef __attribute__((ext_vector_type(8)))  float    v8f;
typedef __attribute__((ext_vector_type(4)))  unsigned int u32x4;
typedef __attribute__((ext_vector_type(8)))  int      i32x8;
typedef __attribute__((ext_vector_type(4)))  int      i32x4;

// ---------------- TDM: load 32x128 f16 tile -> LDS with padded rows ----------------
// D# per CDNA5 ISA: 2D tensor, data_size=1 (2B),
// tile 128(x) x 32(y), tensor_dim0_stride = ldw, pad to BSTR-half rows:
// pad_interval=5 (64 DWORDs = one 128-half row), pad_amount=3 (4 DWORDs = 8 halves).
__device__ __forceinline__ void tdm_load_tile(const _Float16* gsrc,
                                              unsigned ldsoff, int ldw) {
  unsigned long long ga = (unsigned long long)(size_t)gsrc;
  u32x4 g0;
  g0[0] = 1u;                                   // count=1, user mode
  g0[1] = ldsoff;                               // lds_addr
  g0[2] = (unsigned)ga;                         // global_addr[31:0]
  g0[3] = ((unsigned)(ga >> 32) & 0x01FFFFFFu)  // global_addr[56:32]
          | (2u << 30);                         // type=2 (image)
  i32x8 g1;
  g1[0] = (int)((1u << 16)        // data_size = 1 (2 bytes)
              | (1u << 20)        // pad_enable
              | (5u << 22)        // pad_interval: 64 DWORDs
              | (3u << 25));      // pad_amount: 4 DWORDs
  g1[1] = (int)(((unsigned)ldw & 0xFFFFu) << 16);        // tensor_dim0[15:0]
  g1[2] = (int)((((unsigned)ldw >> 16) & 0xFFFFu)        // tensor_dim0[31:16]
              | (0xFFFFu << 16));                        // tensor_dim1[15:0] (large)
  g1[3] = (int)(128u << 16);                             // tile_dim0 = 128
  g1[4] = 32;                                            // tile_dim1 = 32
  g1[5] = ldw;                                           // tensor_dim0_stride[31:0]
  g1[6] = 0;
  g1[7] = 0;
  i32x4 g2 = {};
  i32x4 g3 = {};
#if defined(__clang_major__) && __clang_major__ >= 23
  i32x8 z8 = {};
  __builtin_amdgcn_tensor_load_to_lds(g0, g1, g2, g3, z8, 0);
#else
  __builtin_amdgcn_tensor_load_to_lds(g0, g1, g2, g3, 0);
#endif
}

// Stage a 32 x 128 f16 tile at Bt (LDS offset 0) from src (row stride ldw elems).
__device__ __forceinline__ void stage_b_tile(_Float16* Bt, const _Float16* src,
                                             int ldw, int wave, int tid) {
#if USE_TDM
  if (wave == 0) {
    tdm_load_tile(src, 0u, ldw);
    __builtin_amdgcn_s_wait_tensorcnt(0);
  }
#else
  int r = tid >> 3;             // 0..31
  int c = (tid & 7) * 16;       // 0..112
  const _Float16* s = src + (size_t)r * ldw + c;
  uint4 p0 = *reinterpret_cast<const uint4*>(s);
  uint4 p1 = *reinterpret_cast<const uint4*>(s + 8);
  *reinterpret_cast<uint4*>(&Bt[r * BSTR + c]) = p0;
  *reinterpret_cast<uint4*>(&Bt[r * BSTR + c + 8]) = p1;
#endif
}

// ---------------- WMMA fragment helpers (wave32, 16x16x32 f16) ----------------
// A 16x32 f16: lanes 0-15/16-31 rows M=0..15; elems 0..15 hold K=base+0..15,
// base = (lane>>4)*16.
__device__ __forceinline__ v16h load_a_frag_h(const _Float16* A, int lda,
                                              int row0, int k0, int lane) {
  int m  = row0 + (lane & 15);
  int kb = (lane >> 4) << 4;
  const _Float16* ap = A + (size_t)m * lda + k0 + kb;
  uint4 lo = *reinterpret_cast<const uint4*>(ap);
  uint4 hi = *reinterpret_cast<const uint4*>(ap + 8);
  v16h a;
  __builtin_memcpy(&a, &lo, 16);
  __builtin_memcpy(reinterpret_cast<char*>(&a) + 16, &hi, 16);
  return a;
}

// B 32x16: lane holds column n = lane&15; k = kgroup + elem, kgroup = (lane>>4)*16.
// Contiguous source along k (e.g. K^T with row stride ldb).
__device__ __forceinline__ v16h load_b_frag_cm_h(const _Float16* Km, int ldb,
                                                 int k0, int col, int lane) {
  int n  = col + (lane & 15);
  int kb = k0 + ((lane >> 4) << 4);
  const _Float16* kp = Km + (size_t)n * ldb + kb;
  uint4 lo = *reinterpret_cast<const uint4*>(kp);
  uint4 hi = *reinterpret_cast<const uint4*>(kp + 8);
  v16h b;
  __builtin_memcpy(&b, &lo, 16);
  __builtin_memcpy(reinterpret_cast<char*>(&b) + 16, &hi, 16);
  return b;
}

// B fragment from LDS-staged row-major tile (rows = k, padded stride BSTR).
__device__ __forceinline__ v16h load_b_frag_lds(const _Float16* Bt, int nLocal,
                                                int lane) {
  int kb = (lane >> 4) << 4;
  v16h b;
#pragma unroll
  for (int j = 0; j < 16; ++j) b[j] = Bt[(kb + j) * BSTR + nLocal];
  return b;
}

#define WMMA(a, b, c)                                                         \
  __builtin_amdgcn_wmma_f32_16x16x32_f16(false, (a), false, (b), (short)0,    \
                                         (c), false, false)

// ---------------- GEMM: C = act(A_h[M,K] @ W_h[K,N] + bias) ----------------
template <int ACT, int OUTF16>  // ACT: 0 none, 1 exact GELU
__global__ void wmma_gemm_f16(const _Float16* __restrict__ A,
                              const _Float16* __restrict__ W,
                              const float* __restrict__ bias,
                              float* __restrict__ Cf,
                              _Float16* __restrict__ Ch, int M, int N, int K) {
  __shared__ _Float16 Bt[BK * BSTR];
  int tid  = threadIdx.x;
  int lane = tid & 31;
  int wave = tid >> 5;
  int wm = wave & 1, wn = wave >> 1;
  int row0   = blockIdx.x * BM + wm * 32;
  int col0   = blockIdx.y * BN + wn * 32;
  int n0blk  = blockIdx.y * BN;

  v8f acc[2][2] = {{{}, {}}, {{}, {}}};
  for (int k0 = 0; k0 < K; k0 += BK) {
    stage_b_tile(Bt, W + (size_t)k0 * N + n0blk, N, wave, tid);
    __syncthreads();
    v16h a0 = load_a_frag_h(A, K, row0, k0, lane);
    v16h a1 = load_a_frag_h(A, K, row0 + 16, k0, lane);
    v16h b0 = load_b_frag_lds(Bt, wn * 32 + (lane & 15), lane);
    v16h b1 = load_b_frag_lds(Bt, wn * 32 + 16 + (lane & 15), lane);
    acc[0][0] = WMMA(a0, b0, acc[0][0]);
    acc[0][1] = WMMA(a0, b1, acc[0][1]);
    acc[1][0] = WMMA(a1, b0, acc[1][0]);
    acc[1][1] = WMMA(a1, b1, acc[1][1]);
    __syncthreads();
  }
#pragma unroll
  for (int tm = 0; tm < 2; ++tm)
#pragma unroll
    for (int tn = 0; tn < 2; ++tn) {
      int n  = col0 + tn * 16 + (lane & 15);
      int mb = row0 + tm * 16 + ((lane >> 4) << 3);
      float bv = bias[n];
#pragma unroll
      for (int r = 0; r < 8; ++r) {
        float v = acc[tm][tn][r] + bv;
        if (ACT == 1) v = 0.5f * v * (1.0f + erff(v * 0.7071067811865475f));
        if (OUTF16)
          Ch[(size_t)(mb + r) * N + n] = (_Float16)v;
        else
          Cf[(size_t)(mb + r) * N + n] = v;
      }
    }
}

// ---------------- attention scores: mask(Q Kt / sqrt(DK)) -> f32 d_out ----------------
__global__ void attn_scores(const _Float16* __restrict__ Q,
                            const _Float16* __restrict__ Kb,
                            const int* __restrict__ mask,
                            float* __restrict__ scores) {
  int bh = blockIdx.z;
  int b  = bh >> 3;
  int h  = bh & 7;
  const _Float16* Qp = Q + (size_t)b * Ss * Dm + h * DKk;
  const _Float16* Kp = Kb + (size_t)b * Ss * Dm + h * DKk;
  float* out = scores + (size_t)bh * Ss * Ss;

  int lane = threadIdx.x & 31;
  int wave = threadIdx.x >> 5;
  int wm = wave & 1, wn = wave >> 1;
  int row0 = blockIdx.x * BM + wm * 32;
  int col0 = blockIdx.y * BN + wn * 32;

  v8f acc[2][2] = {{{}, {}}, {{}, {}}};
#pragma unroll
  for (int k0 = 0; k0 < DKk; k0 += BK) {
    v16h a0 = load_a_frag_h(Qp, Dm, row0, k0, lane);
    v16h a1 = load_a_frag_h(Qp, Dm, row0 + 16, k0, lane);
    v16h b0 = load_b_frag_cm_h(Kp, Dm, k0, col0, lane);
    v16h b1 = load_b_frag_cm_h(Kp, Dm, k0, col0 + 16, lane);
    acc[0][0] = WMMA(a0, b0, acc[0][0]);
    acc[0][1] = WMMA(a0, b1, acc[0][1]);
    acc[1][0] = WMMA(a1, b0, acc[1][0]);
    acc[1][1] = WMMA(a1, b1, acc[1][1]);
  }
  const float scale = 0.08838834764831845f;  // 1/sqrt(128)
#pragma unroll
  for (int tm = 0; tm < 2; ++tm)
#pragma unroll
    for (int tn = 0; tn < 2; ++tn) {
      int n  = col0 + tn * 16 + (lane & 15);
      int mb = row0 + tm * 16 + ((lane >> 4) << 3);
      bool neg = (mask[b * Ss + n] == 0);
#pragma unroll
      for (int r = 0; r < 8; ++r)
        out[(size_t)(mb + r) * Ss + n] = neg ? -1e9f : acc[tm][tn][r] * scale;
    }
}

// ---------------- softmax rows (f32 in d_out) + f16 copy for A@V ----------------
__global__ void softmax_dual(float* __restrict__ scores,
                             _Float16* __restrict__ ah) {
  size_t base = (size_t)blockIdx.x * Ss;
  float* row = scores + base;
  _Float16* hrow = ah + base;
  __shared__ float red[256];
  int t = threadIdx.x;
  float a0 = row[t], a1 = row[t + 256];

  red[t] = fmaxf(a0, a1);
  __syncthreads();
  for (int s = 128; s > 0; s >>= 1) {
    if (t < s) red[t] = fmaxf(red[t], red[t + s]);
    __syncthreads();
  }
  float m = red[0];
  __syncthreads();

  float e0 = expf(a0 - m), e1 = expf(a1 - m);
  red[t] = e0 + e1;
  __syncthreads();
  for (int s = 128; s > 0; s >>= 1) {
    if (t < s) red[t] += red[t + s];
    __syncthreads();
  }
  float inv = 1.0f / red[0];
  e0 *= inv;
  e1 *= inv;
  row[t] = e0;
  row[t + 256] = e1;
  hrow[t] = (_Float16)e0;
  hrow[t + 256] = (_Float16)e1;
}

// ---------------- O = A_h[S,S] @ V_h[S,DK] -> f16 in [B,S,D] layout ----------------
__global__ void attn_av(const _Float16* __restrict__ Att,
                        const _Float16* __restrict__ V,
                        _Float16* __restrict__ O) {
  __shared__ _Float16 Bt[BK * BSTR];
  int bh = blockIdx.z;
  int b  = bh >> 3;
  int h  = bh & 7;
  const _Float16* Ap = Att + (size_t)bh * Ss * Ss;
  const _Float16* Vp = V + (size_t)b * Ss * Dm + h * DKk;
  _Float16* Op = O + (size_t)b * Ss * Dm + h * DKk;

  int tid  = threadIdx.x;
  int lane = tid & 31;
  int wave = tid >> 5;
  int wm = wave & 1, wn = wave >> 1;
  int row0 = blockIdx.x * BM + wm * 32;
  int col0 = wn * 32;   // N = 128 = one block tile

  v8f acc[2][2] = {{{}, {}}, {{}, {}}};
  for (int k0 = 0; k0 < Ss; k0 += BK) {
    stage_b_tile(Bt, Vp + (size_t)k0 * Dm, Dm, wave, tid);
    __syncthreads();
    v16h a0 = load_a_frag_h(Ap, Ss, row0, k0, lane);
    v16h a1 = load_a_frag_h(Ap, Ss, row0 + 16, k0, lane);
    v16h b0 = load_b_frag_lds(Bt, wn * 32 + (lane & 15), lane);
    v16h b1 = load_b_frag_lds(Bt, wn * 32 + 16 + (lane & 15), lane);
    acc[0][0] = WMMA(a0, b0, acc[0][0]);
    acc[0][1] = WMMA(a0, b1, acc[0][1]);
    acc[1][0] = WMMA(a1, b0, acc[1][0]);
    acc[1][1] = WMMA(a1, b1, acc[1][1]);
    __syncthreads();
  }
#pragma unroll
  for (int tm = 0; tm < 2; ++tm)
#pragma unroll
    for (int tn = 0; tn < 2; ++tn) {
      int n  = col0 + tn * 16 + (lane & 15);
      int mb = row0 + tm * 16 + ((lane >> 4) << 3);
#pragma unroll
      for (int r = 0; r < 8; ++r)
        Op[(size_t)(mb + r) * Dm + n] = (_Float16)acc[tm][tn][r];
    }
}

// ---------------- dst = LayerNorm(x + r); optional f16 mirror ----------------
__global__ void residual_ln_dual(float* __restrict__ dst,
                                 const float* __restrict__ x,
                                 const float* __restrict__ r,
                                 const float* __restrict__ g,
                                 const float* __restrict__ bta,
                                 _Float16* __restrict__ dh) {
  size_t row = blockIdx.x;
  const float* xp = x + row * Dm;
  const float* rp = r + row * Dm;
  float* dp = dst + row * Dm;
  __shared__ float red[256];
  int t = threadIdx.x;

  float v[4];
  float s = 0.f;
#pragma unroll
  for (int i = 0; i < 4; ++i) {
    v[i] = xp[t + 256 * i] + rp[t + 256 * i];
    s += v[i];
  }
  red[t] = s;
  __syncthreads();
  for (int k = 128; k > 0; k >>= 1) {
    if (t < k) red[t] += red[t + k];
    __syncthreads();
  }
  float mean = red[0] * (1.0f / Dm);
  __syncthreads();

  float vs = 0.f;
#pragma unroll
  for (int i = 0; i < 4; ++i) {
    float d = v[i] - mean;
    vs += d * d;
  }
  red[t] = vs;
  __syncthreads();
  for (int k = 128; k > 0; k >>= 1) {
    if (t < k) red[t] += red[t + k];
    __syncthreads();
  }
  float inv = rsqrtf(red[0] * (1.0f / Dm) + 1e-5f);
  __syncthreads();
#pragma unroll
  for (int i = 0; i < 4; ++i) {
    int d = t + 256 * i;
    float o = (v[i] - mean) * inv * g[d] + bta[d];
    dp[d] = o;
    if (dh) dh[row * Dm + d] = (_Float16)o;
  }
}

// ---------------- embedding + sin/cos PE, dual f32/f16 ----------------
__global__ void embed_pe(const int* __restrict__ ids,
                         const float* __restrict__ emb, float* __restrict__ x,
                         _Float16* __restrict__ xh) {
  int row = blockIdx.x;
  int s   = row & (Ss - 1);
  int id  = ids[row];
  const float c = -9.210340371976184f / (float)Dm;  // -ln(10000)/D
#pragma unroll
  for (int i = 0; i < 4; ++i) {
    int d = threadIdx.x + 256 * i;
    float div = expf((float)((d >> 1) << 1) * c);
    float ang = (float)s * div;
    float pe  = (d & 1) ? cosf(ang) : sinf(ang);
    float o = emb[(size_t)id * Dm + d] + pe;
    x[(size_t)row * Dm + d] = o;
    xh[(size_t)row * Dm + d] = (_Float16)o;
  }
}

// ---------------- f32 -> f16 conversion (for weights) ----------------
__global__ void cvt_f16(const float* __restrict__ src,
                        _Float16* __restrict__ dst, int n) {
  int i = (blockIdx.x * 256 + threadIdx.x) * 4;
  if (i >= n) return;
  float4 v = *reinterpret_cast<const float4*>(src + i);
  dst[i + 0] = (_Float16)v.x;
  dst[i + 1] = (_Float16)v.y;
  dst[i + 2] = (_Float16)v.z;
  dst[i + 3] = (_Float16)v.w;
}

// ---------------- masked mean pool + heads ----------------
__global__ void pool_kernel(const float* __restrict__ x,
                            const int* __restrict__ mask,
                            float* __restrict__ pooled) {
  int b = blockIdx.y;
  int d = blockIdx.x * 256 + threadIdx.x;
  float s = 0.f, ms = 0.f;
  for (int t = 0; t < Ss; ++t) {
    float mf = (float)mask[b * Ss + t];
    s  += x[((size_t)b * Ss + t) * Dm + d] * mf;
    ms += mf;
  }
  pooled[b * Dm + d] = s / ms;
}

__global__ void heads_kernel(const float* __restrict__ pooled,
                             const float* __restrict__ wcls,
                             const float* __restrict__ bcls,
                             const float* __restrict__ wint,
                             const float* __restrict__ bint,
                             float* __restrict__ sent,
                             float* __restrict__ inten) {
  int b = threadIdx.x;
  if (b >= Bb) return;
  const float* p = pooled + b * Dm;
  float l0 = bcls[0], l1 = bcls[1], l2 = bcls[2], li = bint[0];
  for (int d = 0; d < Dm; ++d) {
    float pv = p[d];
    l0 += pv * wcls[d * 3 + 0];
    l1 += pv * wcls[d * 3 + 1];
    l2 += pv * wcls[d * 3 + 2];
    li += pv * wint[d];
  }
  float m = fmaxf(l0, fmaxf(l1, l2));
  float e0 = expf(l0 - m), e1 = expf(l1 - m), e2 = expf(l2 - m);
  float inv = 1.0f / (e0 + e1 + e2);
  sent[b * 3 + 0] = e0 * inv;
  sent[b * 3 + 1] = e1 * inv;
  sent[b * 3 + 2] = e2 * inv;
  inten[b] = 1.0f / (1.0f + expf(-li));
}

// ---------------- orchestration ----------------
extern "C" void kernel_launch(void* const* d_in, const int* in_sizes, int n_in,
                              void* d_out, int out_size, void* d_ws,
                              size_t ws_size, hipStream_t stream) {
  const int*   ids  = (const int*)d_in[0];
  const int*   mask = (const int*)d_in[1];
  const float* emb  = (const float*)d_in[2];
  const float* wq   = (const float*)d_in[3];
  const float* bq   = (const float*)d_in[4];
  const float* wk   = (const float*)d_in[5];
  const float* bk   = (const float*)d_in[6];
  const float* wv   = (const float*)d_in[7];
  const float* bv   = (const float*)d_in[8];
  const float* wo   = (const float*)d_in[9];
  const float* bo   = (const float*)d_in[10];
  const float* w1   = (const float*)d_in[11];
  const float* b1   = (const float*)d_in[12];
  const float* w2   = (const float*)d_in[13];
  const float* b2   = (const float*)d_in[14];
  const float* ln1g = (const float*)d_in[15];
  const float* ln1b = (const float*)d_in[16];
  const float* ln2g = (const float*)d_in[17];
  const float* ln2b = (const float*)d_in[18];
  const float* wcls = (const float*)d_in[19];
  const float* bcls = (const float*)d_in[20];
  const float* wint = (const float*)d_in[21];
  const float* bint = (const float*)d_in[22];

  // d_out (floats): sent[24] | inten[8] | x[4M] | pooled[8192] | attn[4][16M]
  float* out    = (float*)d_out;
  float* sent   = out;
  float* inten  = out + 24;
  float* xout   = out + 32;
  float* pooled = out + 32 + (size_t)NTOK * Dm;
  float* attn0  = pooled + (size_t)Bb * Dm;

  // workspace (~328 MiB): f32: x | oproj(=h2); f16: x_h q_h k_h v_h oin_h a_h h1_h W_h[]
  const size_t NT = (size_t)NTOK * Dm;        // 4,194,304
  const size_t NA = (size_t)Bb * Hh * Ss * Ss;  // 16,777,216
  float* x     = (float*)d_ws;
  float* oproj = x + NT;
  float* h2    = oproj;                        // alias: oproj dead before h2 written
  _Float16* hbase = (_Float16*)(x + 2 * NT);
  _Float16* x_h   = hbase;
  _Float16* q_h   = x_h + NT;
  _Float16* k_h   = q_h + NT;
  _Float16* v_h   = k_h + NT;
  _Float16* oin_h = v_h + NT;
  _Float16* a_h   = oin_h + NT;                // NA halves
  _Float16* h1_h  = a_h + NA;                  // NTOK*FF halves
  _Float16* wq_h  = h1_h + (size_t)NTOK * FFf;
  _Float16* wk_h  = wq_h + Ll * NT;
  _Float16* wv_h  = wk_h + Ll * NT;
  _Float16* wo_h  = wv_h + Ll * NT;
  _Float16* w1_h  = wo_h + Ll * NT;            // L*D*FF
  _Float16* w2_h  = w1_h + (size_t)Ll * Dm * FFf;

  dim3 blk(256);
  auto cvt = [&](const float* s, _Float16* d, size_t n) {
    cvt_f16<<<(unsigned)(n / 1024), blk, 0, stream>>>(s, d, (int)n);
  };
  // one-time f16 weight conversion (192 MiB f16 total: resident in 192 MB L2)
  cvt(wq, wq_h, Ll * NT);
  cvt(wk, wk_h, Ll * NT);
  cvt(wv, wv_h, Ll * NT);
  cvt(wo, wo_h, Ll * NT);
  cvt(w1, w1_h, (size_t)Ll * Dm * FFf);
  cvt(w2, w2_h, (size_t)Ll * FFf * Dm);

  auto gemm = [&](const _Float16* A, const _Float16* W, const float* bias,
                  float* Cf, _Float16* Ch, int M, int N, int K, int act) {
    dim3 g(M / BM, N / BN);
    if (act)
      wmma_gemm_f16<1, 1><<<g, blk, 0, stream>>>(A, W, bias, Cf, Ch, M, N, K);
    else if (Ch)
      wmma_gemm_f16<0, 1><<<g, blk, 0, stream>>>(A, W, bias, Cf, Ch, M, N, K);
    else
      wmma_gemm_f16<0, 0><<<g, blk, 0, stream>>>(A, W, bias, Cf, Ch, M, N, K);
  };

  embed_pe<<<NTOK, blk, 0, stream>>>(ids, emb, x, x_h);

  for (int i = 0; i < Ll; ++i) {
    float* attnL = attn0 + (size_t)i * NA;

    gemm(x_h, wq_h + (size_t)i * NT, bq + i * Dm, nullptr, q_h, NTOK, Dm, Dm, 0);
    gemm(x_h, wk_h + (size_t)i * NT, bk + i * Dm, nullptr, k_h, NTOK, Dm, Dm, 0);
    gemm(x_h, wv_h + (size_t)i * NT, bv + i * Dm, nullptr, v_h, NTOK, Dm, Dm, 0);

    attn_scores<<<dim3(Ss / BM, Ss / BN, Bb * Hh), blk, 0, stream>>>(
        q_h, k_h, mask, attnL);
    softmax_dual<<<Bb * Hh * Ss, blk, 0, stream>>>(attnL, a_h);
    attn_av<<<dim3(Ss / BM, 1, Bb * Hh), blk, 0, stream>>>(a_h, v_h, oin_h);

    gemm(oin_h, wo_h + (size_t)i * NT, bo + i * Dm, oproj, nullptr, NTOK, Dm,
         Dm, 0);
    residual_ln_dual<<<NTOK, blk, 0, stream>>>(x, x, oproj, ln1g + i * Dm,
                                               ln1b + i * Dm, x_h);

    gemm(x_h, w1_h + (size_t)i * Dm * FFf, b1 + i * FFf, nullptr, h1_h, NTOK,
         FFf, Dm, 1);  // + exact GELU, f16 out
    gemm(h1_h, w2_h + (size_t)i * FFf * Dm, b2 + i * Dm, h2, nullptr, NTOK, Dm,
         FFf, 0);

    float* dst = (i == Ll - 1) ? xout : x;
    residual_ln_dual<<<NTOK, blk, 0, stream>>>(
        dst, x, h2, ln2g + i * Dm, ln2b + i * Dm,
        (i == Ll - 1) ? (_Float16*)nullptr : x_h);
  }

  pool_kernel<<<dim3(Dm / 256, Bb), blk, 0, stream>>>(xout, mask, pooled);
  heads_kernel<<<1, 32, 0, stream>>>(pooled, wcls, bcls, wint, bint, sent,
                                     inten);
}
